// ComplexPolarTransformerBeta_17910013625038
// MI455X (gfx1250) — matrix operand order repaired and checked by reading, not compiled, not run on hardware
//
#include <hip/hip_runtime.h>
#include <hip/hip_bf16.h>
#include <math.h>

typedef __attribute__((ext_vector_type(16))) _Float16 v16h;
typedef __attribute__((ext_vector_type(8)))  _Float16 v8h;
typedef __attribute__((ext_vector_type(8)))  float    v8f;
typedef __attribute__((ext_vector_type(4)))  float    v4f;
typedef __attribute__((ext_vector_type(4)))  unsigned int u32x4;
typedef __attribute__((ext_vector_type(8)))  int      i32x8;
typedef __attribute__((ext_vector_type(4)))  int      i32x4;
typedef _Float16 half_t;

#define NATOM  128
#define HID    128
#define NEDGE  1024
#define NLAYER 4
#define LDF    132   // f32 LDS row stride (floats)  -> 528B, 16B aligned, bank-skewed
#define LDH    136   // f16 LDS row stride (halves)  -> 272B, 16B aligned, bank-skewed
#define LDA    264   // f16 row stride for the 128x256 concat buffer
#define LDRP   272   // f16 row stride for staged rp_W (256 + 16 pad halves)

// ---- dynamic LDS layout (bytes) ----
#define OFF_MAGF   0
#define SZ_MAGF    (NATOM*LDF*4)          // 67584
#define OFF_STAGE  (OFF_MAGF+SZ_MAGF)     // weight staging (2 x 34816) | bias f32 (65536)
#define SZ_H       (NATOM*LDH*2)          // 34816
#define SZ_STAGE   (2*SZ_H)               // 69632
#define OFF_PH     (OFF_STAGE+SZ_STAGE)
#define OFF_Q      (OFF_PH+SZ_H)
#define OFF_K      (OFF_Q+SZ_H)
#define OFF_VM     (OFF_K+SZ_H)
#define OFF_VP     (OFF_VM+SZ_H)
#define SMEM_BYTES (OFF_VP+SZ_H)          // 311296 <= 320KB WGP LDS

// ================= TDM (Tensor Data Mover) staging =================
#if defined(__has_builtin)
#  if __has_builtin(__builtin_amdgcn_tensor_load_to_lds) && __has_builtin(__builtin_amdgcn_s_wait_tensorcnt)
#    define HAVE_TDM 1
#  endif
#endif
#ifndef HAVE_TDM
#  define HAVE_TDM 0
#endif

#if HAVE_TDM
#define TDM_WAIT(n) __builtin_amdgcn_s_wait_tensorcnt(n)
// 1D copy of n8 8-byte elements, with LDS padding of 4 DWORDs every 64 DWORDs
// (256B rows -> 272B rows), matching the LDH/LDRP padded strides.
__device__ inline void tdm_issue(const void* src, unsigned lds_off, unsigned n8) {
  unsigned long long ga = (unsigned long long)(uintptr_t)src;
  u32x4 g0;
  g0[0] = 1u;                                      // count=1, user mode, no gather
  g0[1] = lds_off;                                 // lds_addr (bytes)
  g0[2] = (unsigned)(ga & 0xFFFFFFFFu);            // global_addr[31:0]
  g0[3] = (unsigned)((ga >> 32) & 0x1FFFFFFu) | (2u << 30);  // addr[56:32] | type=2
  i32x8 g1;
  g1[0] = (int)((3u << 16) | (1u << 20) | (5u << 22) | (3u << 25));
          // data_size=8B | pad_enable | pad_interval=64dw | pad_amount=4dw
  g1[1] = (int)((n8 & 0xFFFFu) << 16);             // tensor_dim0[15:0]
  g1[2] = (int)((n8 >> 16) | (1u << 16));          // tensor_dim0[31:16] | tensor_dim1=1
  g1[3] = (int)(n8 << 16);                         // tile_dim0 = n8 (fits 16 bits)
  g1[4] = 0;                                       // tile_dim1/2 unused
  g1[5] = (int)n8;                                 // tensor_dim0_stride
  g1[6] = 0; g1[7] = 0;
  i32x4 z4 = {0, 0, 0, 0};
#if __clang_major__ >= 23
  i32x8 z8 = {0, 0, 0, 0, 0, 0, 0, 0};
  __builtin_amdgcn_tensor_load_to_lds(g0, g1, z4, z4, z8, 0);
#else
  __builtin_amdgcn_tensor_load_to_lds(g0, g1, z4, z4, 0);
#endif
}
#else
#define TDM_WAIT(n)
#endif

// Fallback / issue wrapper: called by wave 0 only.
__device__ inline void stage_issue(const half_t* src, char* smembase, unsigned lds_off,
                                   int rows, int rowHalves, int dstLd, int lane) {
#if HAVE_TDM
  (void)dstLd; (void)lane;
  tdm_issue(src, lds_off, (unsigned)(rows * rowHalves) >> 2);
#else
  half_t* dst = (half_t*)(smembase + lds_off);
  const int segs = rowHalves >> 3;
  for (int idx = lane; idx < rows * segs; idx += 32) {
    int r = idx / segs, s = idx - r * segs;
    *(v8h*)(dst + r * dstLd + s * 8) = *(const v8h*)(src + r * rowHalves + s * 8);
  }
#endif
}

// ================= WMMA fragment helpers =================
__device__ inline v8f wmma32(v16h a, v16h b, v8f c) {
  return __builtin_amdgcn_wmma_f32_16x16x32_f16(false, a, false, b, (short)0, c, false, false);
}

// A 16x32 f16 fragment, row-major source. `row` = lane's row base pointer.
__device__ inline v16h loadA_h(const half_t* row, int k0, int hi) {
  const half_t* p0 = row + k0 + hi*8;
  v8h lo = *(const v8h*)p0;
  v8h hg = *(const v8h*)(p0 + 16);
  v16h r;
  #pragma unroll
  for (int i = 0; i < 8; ++i) { r[i] = lo[i]; r[i+8] = hg[i]; }
  return r;
}

// A fragment from f32 source (convert to f16 on the fly).
__device__ inline v16h loadA_f(const float* row, int k0, int hi) {
  const float* p0 = row + k0 + hi*8;
  const float* p1 = p0 + 16;
  v4f a0 = *(const v4f*)p0, a1 = *(const v4f*)(p0+4);
  v4f b0 = *(const v4f*)p1, b1 = *(const v4f*)(p1+4);
  v16h r;
  #pragma unroll
  for (int i = 0; i < 4; ++i) {
    r[i]    = (half_t)a0[i]; r[4+i]  = (half_t)a1[i];
    r[8+i]  = (half_t)b0[i]; r[12+i] = (half_t)b1[i];
  }
  return r;
}

// B 32x16 fragment from "BT" layout (bt[n*ld + k] == B[k][n]).
__device__ inline v16h loadB_h(const half_t* bt, int ld, int col0, int k0, int lane) {
  const half_t* p = bt + (col0 + (lane & 15)) * ld + k0 + ((lane >> 4) << 3);
  v8h lo = *(const v8h*)p;
  v8h hg = *(const v8h*)(p + 16);
  v16h r;
  #pragma unroll
  for (int i = 0; i < 8; ++i) { r[i] = lo[i]; r[i+8] = hg[i]; }
  return r;
}

__device__ inline void init_acc(v8f (&acc)[8], const float* biasvec, int ln16) {
  #pragma unroll
  for (int j = 0; j < 8; ++j) {
    float bv = biasvec ? biasvec[j*16 + ln16] : 0.0f;
    #pragma unroll
    for (int r = 0; r < 8; ++r) acc[j][r] = bv;
  }
}

// K-chunked accumulate: preload 8 B-fragments into distinct registers, then 8
// back-to-back WMMAs, so waits are graduated instead of draining per tile.
template<int KC, bool AF32>
__device__ inline void mm_accum(v8f (&acc)[8], const void* arow_, const half_t* bt,
                                int bld, int lane) {
  const int hi = lane >> 4;
  #pragma unroll
  for (int kc = 0; kc < KC; ++kc) {
    v16h a = AF32 ? loadA_f((const float*)arow_, kc*32, hi)
                  : loadA_h((const half_t*)arow_, kc*32, hi);
    v16h bf[8];
    #pragma unroll
    for (int j = 0; j < 8; ++j) bf[j] = loadB_h(bt, bld, j*16, kc*32, lane);
    #pragma unroll
    for (int j = 0; j < 8; ++j) acc[j] = wmma32(a, bf[j], acc[j]);
  }
}

__device__ inline void store_rowmajor(const v8f (&acc)[8], half_t* dst, int row0, int lane) {
  const int ln16 = lane & 15, hi = lane >> 4;
  #pragma unroll
  for (int j = 0; j < 8; ++j)
    #pragma unroll
    for (int r = 0; r < 8; ++r)
      dst[(row0 + hi*8 + r) * LDH + j*16 + ln16] = (half_t)acc[j][r];
}

__device__ inline void store_trans(const v8f (&acc)[8], half_t* dst, int row0, int lane) {
  const int ln16 = lane & 15, hi = lane >> 4;
  #pragma unroll
  for (int j = 0; j < 8; ++j) {
    v8h tv;
    #pragma unroll
    for (int r = 0; r < 8; ++r) tv[r] = (half_t)acc[j][r];
    *(v8h*)(dst + (j*16 + ln16) * LDH + row0 + hi*8) = tv;  // one b128 store
  }
}

// ================= kernel 1: weight prep (f32 -> f16, transposed) =================
__global__ void prep_weights(const float* Wq, const float* Wk, const float* Wvm,
                             const float* Wvp, const float* rp_W,
                             half_t* wT, half_t* rpT) {
  const int bid = blockIdx.x, tid = threadIdx.x;
  if (bid < 16) {
    const int l = bid >> 2, m = bid & 3;
    const float* src = (m == 0 ? Wq : m == 1 ? Wk : m == 2 ? Wvm : Wvp) + (size_t)l*HID*HID;
    half_t* dst = wT + ((size_t)l*4 + m)*HID*HID;
    for (int idx = tid; idx < HID*HID; idx += 256) {
      int n = idx >> 7, k = idx & 127;
      dst[n*HID + k] = (half_t)src[k*HID + n];
    }
  } else {
    for (int idx = tid; idx < 256*HID; idx += 256) {
      int n = idx >> 8, k = idx & 255;
      rpT[n*256 + k] = (half_t)rp_W[k*HID + n];
    }
  }
}

// ================= kernel 2: per-molecule transformer =================
__global__ __launch_bounds__(256, 1)
void mol_kernel(const float* __restrict__ atom_types, const float* __restrict__ coords,
                const int* __restrict__ edge_index, const float* __restrict__ edge_attr,
                const float* __restrict__ emb_Wm, const float* __restrict__ emb_bm,
                const float* __restrict__ emb_Wp, const float* __restrict__ emb_bp,
                const float* __restrict__ bq, const float* __restrict__ bk,
                const float* __restrict__ bvm, const float* __restrict__ bvp,
                const float* __restrict__ We, const float* __restrict__ be,
                const float* __restrict__ dist_scale,
                const float* __restrict__ ln_g, const float* __restrict__ ln_b,
                const float* __restrict__ rp_b,
                const half_t* __restrict__ wT, const half_t* __restrict__ rpT,
                float* __restrict__ pooled) {
  extern __shared__ char smem[];
  float*  sMag  = (float*)(smem + OFF_MAGF);
  half_t* sStg0 = (half_t*)(smem + OFF_STAGE);          // staged weights, buffer 0
  half_t* sStg1 = (half_t*)(smem + OFF_STAGE + SZ_H);   // staged weights, buffer 1
  float*  sBias = (float*)(smem + OFF_STAGE);           // time-shared with staging
  half_t* sPh   = (half_t*)(smem + OFF_PH);
  half_t* sQ    = (half_t*)(smem + OFF_Q);              // q, then attn
  half_t* sK    = (half_t*)(smem + OFF_K);
  half_t* sVM   = (half_t*)(smem + OFF_VM);
  half_t* sVP   = (half_t*)(smem + OFF_VP);

  const int b    = blockIdx.x;
  const int tid  = threadIdx.x;
  const int lane = tid & 31;
  const int wv   = tid >> 5;
  const int ln16 = lane & 15;
  const int hi   = lane >> 4;
  const int row0 = wv * 16;

  // ---------- embedding ----------
  {
    const float* at = atom_types + (size_t)b * NATOM * 16;
    const float* cs = coords     + (size_t)b * NATOM * 3;
    for (int idx = tid; idx < NATOM*HID; idx += 256) {
      int n = idx >> 7, h = idx & 127;
      float xm = emb_bm[h], xp = emb_bp[h];
      #pragma unroll
      for (int k = 0; k < 16; ++k) {
        float xv = at[n*16 + k];
        xm += xv * emb_Wm[k*HID + h];
        xp += xv * emb_Wp[k*HID + h];
      }
      #pragma unroll
      for (int k = 0; k < 3; ++k) {
        float xv = cs[n*3 + k];
        xm += xv * emb_Wm[(16+k)*HID + h];
        xp += xv * emb_Wp[(16+k)*HID + h];
      }
      sMag[n*LDF + h] = xm;
      sPh[n*LDH + h]  = (half_t)xp;
    }
  }
  __syncthreads();

  for (int l = 0; l < NLAYER; ++l) {
    const half_t* wl    = wT + (size_t)l * 4 * HID * HID;
    const float*  amrow = sMag + (row0 + ln16)*LDF;
    const half_t* aprow = sPh  + (row0 + ln16)*LDH;

    // ---------- stage Wq,Wk (double-buffered TDM), then 4 projection GEMMs ----------
    if (wv == 0) {
      stage_issue(wl + 0*HID*HID, smem, OFF_STAGE,        HID, HID, LDH, lane);
      stage_issue(wl + 1*HID*HID, smem, OFF_STAGE + SZ_H, HID, HID, LDH, lane);
      TDM_WAIT(1);                       // Wq resident
    }
    __syncthreads();
    {
      v8f acc[8]; init_acc(acc, bq + l*HID, ln16);
      mm_accum<4, true>(acc, amrow, sStg0, LDH, lane);
      store_rowmajor(acc, sQ, row0, lane);
    }
    __syncthreads();
    if (wv == 0) {
      stage_issue(wl + 2*HID*HID, smem, OFF_STAGE, HID, HID, LDH, lane);  // Wvm -> buf0
      TDM_WAIT(1);                       // Wk resident
    }
    __syncthreads();
    {
      v8f acc[8]; init_acc(acc, bk + l*HID, ln16);
      mm_accum<4, true>(acc, amrow, sStg1, LDH, lane);
      store_rowmajor(acc, sK, row0, lane);
    }
    __syncthreads();
    if (wv == 0) {
      stage_issue(wl + 3*HID*HID, smem, OFF_STAGE + SZ_H, HID, HID, LDH, lane); // Wvp -> buf1
      TDM_WAIT(1);                       // Wvm resident
    }
    __syncthreads();
    {
      v8f acc[8]; init_acc(acc, bvm + l*HID, ln16);
      mm_accum<4, true>(acc, amrow, sStg0, LDH, lane);
      store_trans(acc, sVM, row0, lane);
    }
    __syncthreads();
    if (wv == 0) TDM_WAIT(0);            // Wvp resident
    __syncthreads();
    {
      v8f acc[8]; init_acc(acc, bvp + l*HID, ln16);
      mm_accum<4, false>(acc, aprow, sStg1, LDH, lane);
      store_trans(acc, sVP, row0, lane);
    }
    __syncthreads();

    // ---------- edge bias (stage region now free) ----------
    for (int idx = tid; idx < NATOM*NATOM; idx += 256) sBias[idx] = 0.0f;
    __syncthreads();
    {
      const int*   ei = edge_index + (size_t)b * 2 * NEDGE;
      const float* ea = edge_attr  + (size_t)b * NEDGE * 4;
      const float w0 = We[l*4+0], w1 = We[l*4+1], w2 = We[l*4+2], w3 = We[l*4+3];
      const float b0 = be[l], dsc = dist_scale[l];
      for (int e = tid; e < NEDGE; e += 256) {
        float a0 = ea[e*4+0], a1 = ea[e*4+1], a2 = ea[e*4+2], a3 = ea[e*4+3];
        float bv = a0*w0 + a1*w1 + a2*w2 + a3*w3 + b0 + dsc*a0;
        atomicAdd(&sBias[ei[e]*NATOM + ei[NEDGE + e]], bv);  // ds_add_f32
      }
    }
    __syncthreads();

    // ---------- scores = (q @ k^T) * scale + bias ; softmax ----------
    {
      const half_t* qrow = sQ + (row0 + ln16)*LDH;
      v8f acc[8]; init_acc(acc, nullptr, ln16);
      mm_accum<4, false>(acc, qrow, sK, LDH, lane);
      const float sc = 0.08838834764831845f;  // 1/sqrt(128)
      #pragma unroll
      for (int j = 0; j < 8; ++j)
        #pragma unroll
        for (int r = 0; r < 8; ++r)
          acc[j][r] = acc[j][r]*sc + sBias[(row0 + hi*8 + r)*NATOM + j*16 + ln16];

      float inv[8];
      #pragma unroll
      for (int r = 0; r < 8; ++r) {      // row m spread over 16 lanes x 8 tiles
        float m = acc[0][r];
        #pragma unroll
        for (int j = 1; j < 8; ++j) m = fmaxf(m, acc[j][r]);
        #pragma unroll
        for (int msk = 8; msk; msk >>= 1) m = fmaxf(m, __shfl_xor(m, msk, 32));
        float s = 0.0f;
        #pragma unroll
        for (int j = 0; j < 8; ++j) { float e = __expf(acc[j][r] - m); acc[j][r] = e; s += e; }
        #pragma unroll
        for (int msk = 8; msk; msk >>= 1) s += __shfl_xor(s, msk, 32);
        inv[r] = 1.0f / s;
      }
      __syncthreads();                    // all waves done reading sQ
      #pragma unroll
      for (int j = 0; j < 8; ++j)
        #pragma unroll
        for (int r = 0; r < 8; ++r)
          sQ[(row0 + hi*8 + r)*LDH + j*16 + ln16] = (half_t)(acc[j][r] * inv[r]);
    }
    __syncthreads();

    // ---------- attn@vm + residual + LayerNorm ; attn@vp + residual ----------
    {
      const half_t* arow = sQ + (row0 + ln16)*LDH;
      float gg[8], bb[8];
      #pragma unroll
      for (int j = 0; j < 8; ++j) {
        gg[j] = ln_g[l*HID + j*16 + ln16];
        bb[j] = ln_b[l*HID + j*16 + ln16];
      }
      v8f acc[8]; init_acc(acc, nullptr, ln16);
      mm_accum<4, false>(acc, arow, sVM, LDH, lane);
      #pragma unroll
      for (int j = 0; j < 8; ++j)
        #pragma unroll
        for (int r = 0; r < 8; ++r)
          acc[j][r] += sMag[(row0 + hi*8 + r)*LDF + j*16 + ln16];   // residual
      #pragma unroll
      for (int r = 0; r < 8; ++r) {
        float s = 0.0f;
        #pragma unroll
        for (int j = 0; j < 8; ++j) s += acc[j][r];
        #pragma unroll
        for (int msk = 8; msk; msk >>= 1) s += __shfl_xor(s, msk, 32);
        float mu = s * (1.0f/HID);
        float v = 0.0f;
        #pragma unroll
        for (int j = 0; j < 8; ++j) { float d = acc[j][r] - mu; v += d*d; }
        #pragma unroll
        for (int msk = 8; msk; msk >>= 1) v += __shfl_xor(v, msk, 32);
        float rinv = rsqrtf(v * (1.0f/HID) + 1e-5f);
        #pragma unroll
        for (int j = 0; j < 8; ++j)
          sMag[(row0 + hi*8 + r)*LDF + j*16 + ln16] =
              gg[j]*(acc[j][r] - mu)*rinv + bb[j];
      }
      v8f accp[8]; init_acc(accp, nullptr, ln16);
      mm_accum<4, false>(accp, arow, sVP, LDH, lane);
      #pragma unroll
      for (int j = 0; j < 8; ++j)
        #pragma unroll
        for (int r = 0; r < 8; ++r) {
          const int m = row0 + hi*8 + r, h = j*16 + ln16;
          sPh[m*LDH + h] = (half_t)(accp[j][r] + (float)sPh[m*LDH + h]);
        }
    }
    __syncthreads();
  }

  // ---------- stage rp_W (overlapped with cos/sin concat) ----------
  if (wv == 0) stage_issue(rpT, smem, OFF_STAGE, HID, 256, LDRP, lane);
  half_t* sA = sK;  // 128 x LDA halves = 67584B, fits in sK+sVM (69632B)
  for (int idx = tid; idx < NATOM*HID; idx += 256) {
    int n = idx >> 7, h = idx & 127;
    float mg = sMag[n*LDF + h];
    float p  = (float)sPh[n*LDH + h];
    float sn, csn; __sincosf(p, &sn, &csn);
    sA[n*LDA + h]       = (half_t)(mg*csn);
    sA[n*LDA + 128 + h] = (half_t)(mg*sn);
  }
  if (wv == 0) TDM_WAIT(0);
  __syncthreads();

  // ---------- atom_repr = A[128x256] @ rp_W + rp_b  (K=256, B from staged LDS) ----------
  {
    const half_t* arow = sA + (row0 + ln16)*LDA;
    v8f acc[8]; init_acc(acc, rp_b, ln16);
    mm_accum<8, false>(acc, arow, sStg0, LDRP, lane);
    #pragma unroll
    for (int j = 0; j < 8; ++j)
      #pragma unroll
      for (int r = 0; r < 8; ++r)
        sMag[(row0 + hi*8 + r)*LDF + j*16 + ln16] = acc[j][r];
  }
  __syncthreads();

  // ---------- pooled = [mean ; sum] over atoms ----------
  if (tid < HID) {
    float s = 0.0f;
    for (int m = 0; m < NATOM; ++m) s += sMag[m*LDF + tid];
    pooled[(size_t)b*256 + tid]       = s * (1.0f/NATOM);
    pooled[(size_t)b*256 + 128 + tid] = s;
  }
}

// ================= kernel 3: output head =================
__global__ __launch_bounds__(128)
void head_kernel(const float* __restrict__ pooled,
                 const float* __restrict__ h1_W, const float* __restrict__ h1_b,
                 const float* __restrict__ h2_W, const float* __restrict__ h2_b,
                 float* __restrict__ out) {
  __shared__ float red[128];
  const int b = blockIdx.x, h = threadIdx.x;
  const float* p = pooled + (size_t)b*256;
  float d = h1_b[h];
  for (int k = 0; k < 256; ++k) d += p[k] * h1_W[k*128 + h];
  float s = d / (1.0f + __expf(-d));   // SiLU
  red[h] = s * h2_W[h];
  __syncthreads();
  for (int off = 64; off > 0; off >>= 1) {
    if (h < off) red[h] += red[h + off];
    __syncthreads();
  }
  if (h == 0) out[b] = red[0] + h2_b[0];
}

// ================= launch =================
extern "C" void kernel_launch(void* const* d_in, const int* in_sizes, int n_in,
                              void* d_out, int out_size, void* d_ws, size_t ws_size,
                              hipStream_t stream) {
  const float* atom_types = (const float*)d_in[0];
  const float* coords     = (const float*)d_in[1];
  const int*   edge_index = (const int*)  d_in[2];
  const float* edge_attr  = (const float*)d_in[3];
  const float* emb_Wm = (const float*)d_in[4];
  const float* emb_bm = (const float*)d_in[5];
  const float* emb_Wp = (const float*)d_in[6];
  const float* emb_bp = (const float*)d_in[7];
  const float* Wq  = (const float*)d_in[8];   const float* bq  = (const float*)d_in[9];
  const float* Wk  = (const float*)d_in[10];  const float* bk  = (const float*)d_in[11];
  const float* Wvm = (const float*)d_in[12];  const float* bvm = (const float*)d_in[13];
  const float* Wvp = (const float*)d_in[14];  const float* bvp = (const float*)d_in[15];
  const float* We  = (const float*)d_in[16];  const float* be  = (const float*)d_in[17];
  const float* dist_scale = (const float*)d_in[18];
  const float* ln_g = (const float*)d_in[19]; const float* ln_b = (const float*)d_in[20];
  const float* rp_W = (const float*)d_in[21]; const float* rp_b = (const float*)d_in[22];
  const float* h1_W = (const float*)d_in[23]; const float* h1_b = (const float*)d_in[24];
  const float* h2_W = (const float*)d_in[25]; const float* h2_b = (const float*)d_in[26];
  float* out = (float*)d_out;

  // workspace layout
  half_t* wT     = (half_t*)d_ws;                         // 16 * 128*128 f16 = 512KB
  half_t* rpT    = wT + (size_t)16*HID*HID;               // 128*256 f16      = 64KB
  float*  pooled = (float*)((char*)d_ws + 524288 + 65536);// 1024*256 f32     = 1MB

  (void)hipFuncSetAttribute((const void*)mol_kernel,
                            hipFuncAttributeMaxDynamicSharedMemorySize, SMEM_BYTES);

  prep_weights<<<17, 256, 0, stream>>>(Wq, Wk, Wvm, Wvp, rp_W, wT, rpT);
  mol_kernel<<<1024, 256, SMEM_BYTES, stream>>>(
      atom_types, coords, edge_index, edge_attr,
      emb_Wm, emb_bm, emb_Wp, emb_bp,
      bq, bk, bvm, bvp, We, be, dist_scale, ln_g, ln_b, rp_b,
      wT, rpT, pooled);
  head_kernel<<<1024, 128, 0, stream>>>(pooled, h1_W, h1_b, h2_W, h2_b, out);
}
